// GatheringLoss_37529424233256
// MI455X (gfx1250) — matrix operand order repaired and checked by reading, not compiled, not run on hardware
//
#include <hip/hip_runtime.h>
#include <hip/hip_bf16.h>

// Problem constants (reference: queries (8,4096,512) f32, items (1024,512) f32)
#define BD   8
#define SD   4096
#define FD   512
#define KD   1024
#define LOG2S 12
#define PI_F 3.14159265358979323846f

typedef __attribute__((ext_vector_type(8)))  float  v8f;
typedef __attribute__((ext_vector_type(16))) __bf16 v16bf;

union Frag16 {
    v16bf v;
    uint4 q[2];
};

__device__ __forceinline__ unsigned short f2bf(float x) {
    unsigned int u = __float_as_uint(x);
    unsigned int r = u + 0x7fffu + ((u >> 16) & 1u); // round-to-nearest-even
    return (unsigned short)(r >> 16);
}

// ---------------------------------------------------------------------------
// Kernel 0: one-shot items f32 -> bf16 conversion (1 MB, stays L2 resident).
// ---------------------------------------------------------------------------
__global__ void __launch_bounds__(256)
convert_items_kernel(const float* __restrict__ items,
                     unsigned short* __restrict__ items_bf16) {
    int i = blockIdx.x * 256 + threadIdx.x;          // float4 index, 131072 total
    float4 v = ((const float4*)items)[i];
    unsigned long long pk =
          (unsigned long long)f2bf(v.x)
        | ((unsigned long long)f2bf(v.y) << 16)
        | ((unsigned long long)f2bf(v.z) << 32)
        | ((unsigned long long)f2bf(v.w) << 48);
    ((unsigned long long*)items_bf16)[i] = pk;
}

// ---------------------------------------------------------------------------
// Kernel 1: per (b, feature-pair) column, compute
//   unit = irfft(exp(-i*angle(rfft(q)))) along the sequence axis,
// via one packed complex FFT (two real columns as re/im), Hermitian split,
// phase extraction, inverse FFT.  Output written as bf16 row-major (M=B*S, F).
// ---------------------------------------------------------------------------
__global__ void __launch_bounds__(256)
fft_phase_kernel(const float* __restrict__ q, unsigned short* __restrict__ unit_bf16) {
    __shared__ float2 z[SD];                     // 32 KB complex workspace

    const int t     = threadIdx.x;               // 0..255
    const int blk   = blockIdx.x;                // 0..(BD*FD/2 - 1)
    const int b     = blk >> 8;                  // FD/2 = 256 pairs per batch
    const int fpair = blk & 255;
    const int f0    = fpair << 1;

    // Load two adjacent feature columns packed as complex (float2, 8B aligned).
    const float2* qcol = (const float2*)(q + (size_t)b * SD * FD + f0);
    #pragma unroll
    for (int j = 0; j < 16; ++j) {
        int s = t + 256 * j;
        z[s] = qcol[(size_t)s * (FD / 2)];
    }
    __syncthreads();

    // Forward FFT, radix-2 DIF (natural in -> bit-reversed out).
    for (int sh = LOG2S - 1; sh >= 0; --sh) {
        const int span = 1 << sh;
        #pragma unroll
        for (int j = 0; j < 8; ++j) {
            int bf  = t + 256 * j;               // butterfly id 0..2047
            int grp = bf >> sh;
            int jj  = bf & (span - 1);
            int i0  = (grp << (sh + 1)) + jj;
            float ang = PI_F * (float)jj / (float)span;
            float sn, cs;
            __sincosf(ang, &sn, &cs);            // w = cs - i*sn  (e^{-i*ang})
            float2 a  = z[i0];
            float2 bb = z[i0 + span];
            float2 s0 = { a.x + bb.x, a.y + bb.y };
            float2 d0 = { a.x - bb.x, a.y - bb.y };
            z[i0]        = s0;
            z[i0 + span] = make_float2(d0.x * cs + d0.y * sn,
                                       d0.y * cs - d0.x * sn);
        }
        __syncthreads();
    }

    // Pointwise phase-only op in the bit-reversed domain.
    float2 G[16];
    #pragma unroll
    for (int j = 0; j < 16; ++j) {
        int p  = t + 256 * j;
        int k  = (int)(__brev((unsigned)p) >> (32 - LOG2S));
        int kn = (SD - k) & (SD - 1);
        int pq = (int)(__brev((unsigned)kn) >> (32 - LOG2S));
        float2 Zp = z[p];
        float2 Zq = z[pq];
        // F1[k] = (Z[k] + conj(Z[S-k]))/2 ; F2[k] = -i*(Z[k] - conj(Z[S-k]))/2
        float f1r = 0.5f * (Zp.x + Zq.x);
        float f1i = 0.5f * (Zp.y - Zq.y);
        float f2r = 0.5f * (Zp.y + Zq.y);
        float f2i = 0.5f * (Zq.x - Zp.x);
        // g = conj(F)/|F|  (angle(0)=0 -> g=1, matching numpy)
        float m1 = f1r * f1r + f1i * f1i;
        float m2 = f2r * f2r + f2i * f2i;
        float g1r = 1.0f, g1i = 0.0f, g2r = 1.0f, g2i = 0.0f;
        if (m1 > 0.0f) { float inv = rsqrtf(m1); g1r = f1r * inv; g1i = -f1i * inv; }
        if (m2 > 0.0f) { float inv = rsqrtf(m2); g2r = f2r * inv; g2i = -f2i * inv; }
        // Repack: G = g1 + i*g2
        G[j].x = g1r - g2i;
        G[j].y = g1i + g2r;
    }
    __syncthreads();
    #pragma unroll
    for (int j = 0; j < 16; ++j) z[t + 256 * j] = G[j];
    __syncthreads();

    // Inverse FFT, radix-2 DIT (bit-reversed in -> natural out).
    for (int sh = 0; sh < LOG2S; ++sh) {
        const int span = 1 << sh;
        #pragma unroll
        for (int j = 0; j < 8; ++j) {
            int bf  = t + 256 * j;
            int grp = bf >> sh;
            int jj  = bf & (span - 1);
            int i0  = (grp << (sh + 1)) + jj;
            float ang = PI_F * (float)jj / (float)span;
            float sn, cs;
            __sincosf(ang, &sn, &cs);            // w = cs + i*sn  (e^{+i*ang})
            float2 a  = z[i0];
            float2 bb = z[i0 + span];
            float2 tb = { bb.x * cs - bb.y * sn, bb.x * sn + bb.y * cs };
            z[i0]        = make_float2(a.x + tb.x, a.y + tb.y);
            z[i0 + span] = make_float2(a.x - tb.x, a.y - tb.y);
        }
        __syncthreads();
    }

    // Scale by 1/S, unpack the two real columns, store bf16 pair (one dword).
    const float invS = 1.0f / (float)SD;
    unsigned int* uw = (unsigned int*)unit_bf16;
    #pragma unroll
    for (int j = 0; j < 16; ++j) {
        int s = t + 256 * j;
        float2 u = z[s];
        unsigned int packed = (unsigned int)f2bf(u.x * invS)
                            | ((unsigned int)f2bf(u.y * invS) << 16);
        uw[(size_t)(b * SD + s) * (FD / 2) + fpair] = packed;
    }
}

// ---------------------------------------------------------------------------
// Async strip staging: copy one 16x512 bf16 items strip (16 KB, contiguous)
// from global to LDS via CDNA5 async-copy (ASYNCcnt-tracked), 256 threads
// x 16 B x 4 issues.
// ---------------------------------------------------------------------------
__device__ __forceinline__ void stage_strip_async(const unsigned short* gsrc,
                                                  unsigned short* sdst, int tid) {
    unsigned    l0 = (unsigned)(size_t)sdst + (unsigned)(tid * 16);
    const char* g0 = (const char*)gsrc + tid * 16;
    #pragma unroll
    for (int i = 0; i < 4; ++i) {
        unsigned    l = l0 + i * 4096;
        const char* g = g0 + i * 4096;
        asm volatile("global_load_async_to_lds_b128 %0, %1, off"
                     :: "v"(l), "v"(g) : "memory");
    }
}

// ---------------------------------------------------------------------------
// Kernel 2: fused  score = unit @ items^T  (WMMA bf16, f32 acc),
//           idx = argmax_k score,  loss = sum_f (q - items[idx])^2.
// Block = 256 threads = 8 waves; each wave owns a 16-row M tile (all K=512 A
// fragments preloaded in VGPRs); items strips are double-buffered in LDS via
// async copies, and B fragments are double-buffered in registers inside the
// K loop so ds_load latency overlaps the WMMA pipe.
// ---------------------------------------------------------------------------
__global__ void __launch_bounds__(256)
gemm_argmax_loss_kernel(const float* __restrict__ q,
                        const float* __restrict__ items,
                        const unsigned short* __restrict__ unit_bf16,
                        const unsigned short* __restrict__ items_bf16,
                        float* __restrict__ out) {
    __shared__ __align__(16) unsigned short sB[2 * 16 * FD];  // 2 x 16 KB strips
    __shared__ int sIdx[8 * 16];

    const int tid   = threadIdx.x;
    const int wave  = tid >> 5;
    const int lane  = tid & 31;
    const int mw    = blockIdx.x * 128 + wave * 16;   // wave's first row
    const int mlane = mw + (lane & 15);
    const int khalf = lane >> 4;                 // 0: K 0-7/16-23, 1: K 8-15/24-31

    // Preload all 16 A fragments (16x32 bf16 each) for this wave's 16 rows.
    // A layout (ISA 7.12.2, 16-bit 16x32): lane m = lane&15;
    //   u16[0..7]  -> K = khalf*8 + j,  u16[8..15] -> K = 16 + khalf*8 + (j-8)
    Frag16 afr[16];
    const unsigned short* arow = unit_bf16 + (size_t)mlane * FD;
    #pragma unroll
    for (int ks = 0; ks < 16; ++ks) {
        int k0 = ks * 32 + khalf * 8;
        afr[ks].q[0] = *(const uint4*)(arow + k0);
        afr[ks].q[1] = *(const uint4*)(arow + k0 + 16);
    }

    float bestVal[8];
    int   bestIdx[8];
    #pragma unroll
    for (int r = 0; r < 8; ++r) { bestVal[r] = -3.402823466e+38f; bestIdx[r] = 0; }

    const int nB = lane & 15;                    // this lane's column in the tile

    // Prime the async pipeline with strip 0.
    stage_strip_async(items_bf16, &sB[0], tid);

    for (int nt = 0; nt < KD / 16; ++nt) {
        // Issue next strip into the other buffer, then wait for the current
        // strip (in-order completion: <=4 outstanding leaves only the newest).
        if (nt < KD / 16 - 1) {
            stage_strip_async(items_bf16 + (size_t)(nt + 1) * 16 * FD,
                              &sB[((nt + 1) & 1) * 16 * FD], tid);
            asm volatile("s_wait_asynccnt 0x4" ::: "memory");
        } else {
            asm volatile("s_wait_asynccnt 0x0" ::: "memory");
        }
        __syncthreads();                         // whole strip visible block-wide

        // 16 K-steps of v_wmma_f32_16x16x32_bf16, B frag double-buffered.
        // B layout (same as A): u16[0..7] -> K = khalf*8 + j,
        // u16[8..15] -> K = 16 + khalf*8 + (j-8); N = lane&15.
        v8f c = {};
        const unsigned short* brow = &sB[(nt & 1) * 16 * FD + nB * FD];
        Frag16 bcur;
        bcur.q[0] = *(const uint4*)(brow + khalf * 8);
        bcur.q[1] = *(const uint4*)(brow + khalf * 8 + 16);
        #pragma unroll
        for (int ks = 0; ks < 16; ++ks) {
            Frag16 bnext;
            if (ks < 15) {
                int k0 = (ks + 1) * 32 + khalf * 8;
                bnext.q[0] = *(const uint4*)(brow + k0);
                bnext.q[1] = *(const uint4*)(brow + k0 + 16);
            }
            c = __builtin_amdgcn_wmma_f32_16x16x32_bf16(
                    false, afr[ks].v, false, bcur.v,
                    (short)0, c, false, false);
            if (ks < 15) bcur = bnext;
        }

        // Running argmax: C layout puts row r (lanes 0-15) / row 8+r (16-31)
        // in VGPR r; column = lane&15 within this N tile.
        #pragma unroll
        for (int r = 0; r < 8; ++r) {
            if (c[r] > bestVal[r]) { bestVal[r] = c[r]; bestIdx[r] = nt * 16 + nB; }
        }
        __syncthreads();                         // reads done before buffer reuse
    }

    // Cross-lane argmax over the 16 columns; xor masks < 16 keep the two
    // row-halves (lanes 0-15 vs 16-31) independent. Prefer lower index on tie
    // to match jnp.argmax first-occurrence semantics.
    #pragma unroll
    for (int m = 1; m < 16; m <<= 1) {
        #pragma unroll
        for (int r = 0; r < 8; ++r) {
            float ov = __shfl_xor(bestVal[r], m, 32);
            int   oi = __shfl_xor(bestIdx[r], m, 32);
            if (ov > bestVal[r] || (ov == bestVal[r] && oi < bestIdx[r])) {
                bestVal[r] = ov; bestIdx[r] = oi;
            }
        }
    }
    if ((lane & 15) == 0) {
        int rowbase = khalf * 8;
        #pragma unroll
        for (int r = 0; r < 8; ++r) sIdx[wave * 16 + rowbase + r] = bestIdx[r];
    }
    __syncthreads();

    // Fused gathering loss: loss[m] = sum_f (q[m,f] - items[idx[m],f])^2.
    // Two lanes per row, interleaved float4 chunks, combine via shuffle.
    const int row  = lane >> 1;
    const int half = lane & 1;
    const int mg   = mw + row;
    const int ib   = sIdx[wave * 16 + row];
    const float4* qr = (const float4*)(q + (size_t)mg * FD);
    const float4* ir = (const float4*)(items + (size_t)ib * FD);
    float acc = 0.0f;
    for (int c4 = half; c4 < FD / 4; c4 += 2) {
        float4 qv = qr[c4];
        float4 iv = ir[c4];
        float dx = qv.x - iv.x, dy = qv.y - iv.y;
        float dz = qv.z - iv.z, dw = qv.w - iv.w;
        acc += dx * dx + dy * dy + dz * dz + dw * dw;
    }
    acc += __shfl_xor(acc, 1, 32);
    if (half == 0) out[mg] = acc;
}

// ---------------------------------------------------------------------------
extern "C" void kernel_launch(void* const* d_in, const int* in_sizes, int n_in,
                              void* d_out, int out_size, void* d_ws, size_t ws_size,
                              hipStream_t stream) {
    (void)in_sizes; (void)n_in; (void)out_size; (void)ws_size;
    const float* queries = (const float*)d_in[0];   // (8, 4096, 512) f32
    const float* items   = (const float*)d_in[1];   // (1024, 512)   f32
    unsigned short* unit = (unsigned short*)d_ws;              // 32 MB bf16 scratch
    unsigned short* items_bf16 =
        (unsigned short*)((char*)d_ws + (size_t)BD * SD * FD * 2); // +1 MB
    float* out           = (float*)d_out;           // (8, 4096)     f32

    // 0) one-shot items f32 -> bf16 (L2 resident thereafter)
    convert_items_kernel<<<(KD * FD / 4) / 256, 256, 0, stream>>>(items, items_bf16);
    // 1) phase-only reconstruction along S (2 feature columns per block)
    fft_phase_kernel<<<BD * (FD / 2), 256, 0, stream>>>(queries, unit);
    // 2) WMMA GEMM + argmax + gathering loss, async-pipelined items strips
    gemm_argmax_loss_kernel<<<(BD * SD) / 128, 256, 0, stream>>>(
        queries, items, unit, items_bf16, out);
}
